// KMeanLayer_35175782154734
// MI455X (gfx1250) — matrix operand order repaired and checked by reading, not compiled, not run on hardware
//
#include <hip/hip_runtime.h>

#define NPTS 131072
#define KC   2048
#define DIM  256

typedef __attribute__((ext_vector_type(16))) __bf16 v16bf;
typedef __attribute__((ext_vector_type(8)))  float  v8f;
typedef __attribute__((ext_vector_type(4)))  int    v4i;
typedef __attribute__((ext_vector_type(4)))  float  v4f;

struct QPair { v4i a, b; };   // 32 bytes == one v16bf operand chunk

#if __has_builtin(__builtin_amdgcn_global_load_async_to_lds_b128)
#define HAVE_ASYNC_LDS 1
typedef __attribute__((address_space(1))) v4i g_v4i;   // global
typedef __attribute__((address_space(3))) v4i l_v4i;   // LDS
static __device__ __forceinline__ void async_copy16(const void* g, void* l) {
  __builtin_amdgcn_global_load_async_to_lds_b128((g_v4i*)g, (l_v4i*)l, 0, 0);
}
#else
#define HAVE_ASYNC_LDS 0
#endif

static __device__ __forceinline__ void wait_asynccnt0() {
#if __has_builtin(__builtin_amdgcn_s_wait_asynccnt)
  __builtin_amdgcn_s_wait_asynccnt(0);
#else
  asm volatile("s_wait_asynccnt 0" ::: "memory");
#endif
}

// ---------------------------------------------------------------------------
// Kernel 1: preprocess centroids.
//   Writes Abuf: per 16-centroid tile kt (128 tiles), the (-2*c) values split
//   into bf16 hi/lo, laid out exactly as the WMMA 16-bit A-operand (16x32 MxK
//   chunk per 32 dims): tile = 16 KB (hi 8 KB then lo 8 KB).
//   A layout (dense 16-bit A 16x32): lane L: m = L%16, half = L/16;
//   element j: d = kbase + half*8 + j        (j<8)
//              d = kbase + 16 + half*8 + j-8 (j>=8)
// ---------------------------------------------------------------------------
__global__ void prep_centroids(const float* __restrict__ cen,
                               __bf16* __restrict__ Abuf) {
  const int kt = blockIdx.x;        // 0..127
  const int t  = threadIdx.x;       // 0..255
  const int c  = t >> 5;            // dim-chunk 0..7
  const int L  = t & 31;            // lane slot
  const int m  = kt * 16 + (L & 15);
  const int half = L >> 4;
  __bf16* hi = Abuf + (size_t)kt * 8192 + (c * 32 + L) * 16;
  __bf16* lo = hi + 4096;
  #pragma unroll
  for (int j = 0; j < 16; ++j) {
    const int d = c * 32 + half * 8 + j + (j >= 8 ? 8 : 0);
    const float v = -2.0f * cen[(size_t)m * DIM + d];
    const __bf16 h = (__bf16)v;
    const float  r = v - (float)h;
    hi[j] = h;
    lo[j] = (__bf16)r;
  }
}

// Kernel 2: per-centroid squared norms.
__global__ void centroid_norms(const float* __restrict__ cen,
                               float* __restrict__ c2) {
  const int i = blockIdx.x * blockDim.x + threadIdx.x;   // 0..2047
  const v4f* row = (const v4f*)(cen + (size_t)i * DIM);
  float s = 0.0f;
  #pragma unroll 8
  for (int d = 0; d < DIM / 4; ++d) {
    v4f f = row[d];
    s += f[0]*f[0] + f[1]*f[1] + f[2]*f[2] + f[3]*f[3];
  }
  c2[i] = s;
}

// ---------------------------------------------------------------------------
// Kernel 3: main assignment. 256 threads = 8 waves; 128 points per block.
// ---------------------------------------------------------------------------
__global__ void __launch_bounds__(256)
assign_points(const float* __restrict__ x, const float* __restrict__ cen,
              const float* __restrict__ c2, const v4i* __restrict__ AbufQ,
              int* __restrict__ ids, float* __restrict__ xc) {
  __shared__ v4i lds[2][1024];           // 2 x 16 KB A-tile double buffer

  const int t    = threadIdx.x;
  const int wave = t >> 5;
  const int lane = t & 31;
  const int half = lane >> 4;
  const int n    = lane & 15;
  const int pointBase = blockIdx.x * 128 + wave * 16;
  const int p = pointBase + n;

  // ---- load this lane's point into B operands, split fp32 -> bf16 hi/lo
  // B layout (32x16 KxN): lane L: n = L%16, halfK = L/16;
  //   element j -> d = 32*cc + halfK*8 + j (j<8), 32*cc + 16 + halfK*8 + j-8 (j>=8)
  v16bf bh[8], bl[8];
  #pragma unroll
  for (int cc = 0; cc < 8; ++cc) {
    const v4f* xp = (const v4f*)(x + (size_t)p * DIM + cc * 32 + half * 8);
    float xv[16];
    #pragma unroll
    for (int q = 0; q < 4; ++q) {
      v4f f = xp[q + (q >= 2 ? 2 : 0)];
      xv[q*4+0] = f[0]; xv[q*4+1] = f[1]; xv[q*4+2] = f[2]; xv[q*4+3] = f[3];
    }
    #pragma unroll
    for (int j = 0; j < 16; ++j) {
      const float v = xv[j];
      const __bf16 h = (__bf16)v;
      bh[cc][j] = h;
      bl[cc][j] = (__bf16)(v - (float)h);
    }
  }

  float best = 3.4e38f;
  int bestIdx = 0;

#if HAVE_ASYNC_LDS
  // ---- async staging: memory -> LDS without touching VGPRs (ASYNCcnt)
  {
    const char* gsrc = (const char*)AbufQ + t * 16;
    char* ldst = (char*)&lds[0][0] + t * 16;
    #pragma unroll
    for (int i = 0; i < 4; ++i)
      async_copy16(gsrc + i * 4096, ldst + i * 4096);
  }
  wait_asynccnt0();
  __syncthreads();
#else
  v4i pf[4];
  #pragma unroll
  for (int i = 0; i < 4; ++i) pf[i] = AbufQ[i * 256 + t];
  #pragma unroll
  for (int i = 0; i < 4; ++i) lds[0][i * 256 + t] = pf[i];
  __syncthreads();
#endif

  #pragma unroll 1
  for (int kt = 0; kt < 128; ++kt) {
    const int buf = kt & 1;

#if HAVE_ASYNC_LDS
    // issue async staging of the next tile into the other buffer
    if (kt + 1 < 128) {
      const char* gs = (const char*)AbufQ + (size_t)(kt + 1) * 16384 + t * 16;
      char* ld = (char*)&lds[buf ^ 1][0] + t * 16;
      #pragma unroll
      for (int i = 0; i < 4; ++i)
        async_copy16(gs + i * 4096, ld + i * 4096);
    }
#else
    if (kt + 1 < 128) {
      #pragma unroll
      for (int i = 0; i < 4; ++i)
        pf[i] = AbufQ[(size_t)(kt + 1) * 1024 + i * 256 + t];
    }
#endif

    // ---- 24 WMMAs, software-pipelined A-chunk (hi/lo) register double buffer
    v8f a0 = {}, a1 = {}, a2 = {};
    const v4i* Lq = lds[buf];
    v4i h0a = Lq[lane * 2],       h0b = Lq[lane * 2 + 1];
    v4i l0a = Lq[lane * 2 + 512], l0b = Lq[lane * 2 + 513];
    #pragma unroll
    for (int cc = 0; cc < 8; ++cc) {
      v4i h1a, h1b, l1a, l1b;
      if (cc < 7) {                       // prefetch next chunk while computing
        const int bj = ((cc + 1) * 32 + lane) * 2;
        h1a = Lq[bj];       h1b = Lq[bj + 1];
        l1a = Lq[bj + 512]; l1b = Lq[bj + 513];
      }
      QPair qh; qh.a = h0a; qh.b = h0b;
      QPair ql; ql.a = l0a; ql.b = l0b;
      const v16bf ah = __builtin_bit_cast(v16bf, qh);
      const v16bf al = __builtin_bit_cast(v16bf, ql);
      a0 = __builtin_amdgcn_wmma_f32_16x16x32_bf16(false, ah, false, bh[cc],
                                                   (short)0, a0, false, false);
      a1 = __builtin_amdgcn_wmma_f32_16x16x32_bf16(false, ah, false, bl[cc],
                                                   (short)0, a1, false, false);
      a2 = __builtin_amdgcn_wmma_f32_16x16x32_bf16(false, al, false, bh[cc],
                                                   (short)0, a2, false, false);
      if (cc < 7) { h0a = h1a; h0b = h1b; l0a = l1a; l0b = l1b; }
    }

    // ---- epilogue: score = dot(-2c, x) + ||c||^2, update running argmin
    const v4f* cv = (const v4f*)(c2 + kt * 16 + half * 8);
    const v4f cv0 = cv[0], cv1 = cv[1];
    const float cva[8] = {cv0[0], cv0[1], cv0[2], cv0[3],
                          cv1[0], cv1[1], cv1[2], cv1[3]};
    #pragma unroll
    for (int i = 0; i < 8; ++i) {
      const float s = a0[i] + a1[i] + a2[i] + cva[i];
      const int idx = kt * 16 + half * 8 + i;
      if (s < best) { best = s; bestIdx = idx; }
    }

#if HAVE_ASYNC_LDS
    wait_asynccnt0();       // this wave's async writes to buf^1 are in LDS
    __syncthreads();        // all waves done: next tile visible, buf free
#else
    __syncthreads();        // everyone done reading lds[buf^1]'s old tile
    if (kt + 1 < 128) {
      #pragma unroll
      for (int i = 0; i < 4; ++i) lds[buf ^ 1][i * 256 + t] = pf[i];
    }
    __syncthreads();        // next tile visible
#endif
  }

  // ---- combine the two centroid-halves held by lane and lane^16
  const float ob = __shfl_xor(best, 16);
  const int   oi = __shfl_xor(bestIdx, 16);
  if (ob < best || (ob == best && oi < bestIdx)) { best = ob; bestIdx = oi; }

  if (lane < 16) ids[p] = bestIdx;

  // ---- wave-cooperative gather of the winning centroid rows (fp32 exact)
  #pragma unroll 1
  for (int pt = 0; pt < 16; ++pt) {
    const int cidx = __shfl(bestIdx, pt);
    const v4f* src = (const v4f*)(cen + (size_t)cidx * DIM);
    v4f* dst = (v4f*)(xc + (size_t)(pointBase + pt) * DIM);
    dst[lane * 2]     = src[lane * 2];
    dst[lane * 2 + 1] = src[lane * 2 + 1];
  }
}

// ---------------------------------------------------------------------------
extern "C" void kernel_launch(void* const* d_in, const int* in_sizes, int n_in,
                              void* d_out, int out_size, void* d_ws, size_t ws_size,
                              hipStream_t stream) {
  const float* x   = (const float*)d_in[0];   // [N, D] fp32
  const float* cen = (const float*)d_in[1];   // [K, D] fp32

  // workspace: 2 MB bf16 hi/lo tiled centroids + 8 KB norms
  __bf16* Abuf = (__bf16*)d_ws;
  float*  c2   = (float*)((char*)d_ws + (size_t)2 * 1024 * 1024);

  int*   ids = (int*)d_out;               // first N slots: int32 ids
  float* xc  = (float*)d_out + NPTS;      // then N*D fp32 gathered centroids

  prep_centroids<<<KC / 16, 256, 0, stream>>>(cen, Abuf);
  centroid_norms<<<KC / 256, 256, 0, stream>>>(cen, c2);
  assign_points<<<NPTS / 128, 256, 0, stream>>>(x, cen, c2, (const v4i*)d_ws,
                                                ids, xc);
}
